// TimeMixing_13692355739857
// MI455X (gfx1250) — matrix operand order repaired and checked by reading, not compiled, not run on hardware
//
#include <hip/hip_runtime.h>
#include <hip/hip_bf16.h>

typedef __attribute__((ext_vector_type(16))) __bf16 v16bf;
typedef __attribute__((ext_vector_type(8)))  __bf16 v8bf;
typedef __attribute__((ext_vector_type(8)))  float  v8f;

static constexpr int Bb   = 8;
static constexpr int Tt   = 2048;
static constexpr int Cc   = 1024;
static constexpr int Mtot = Bb * Tt;       // 16384 rows

// ---------------------------------------------------------------------------
// fp32 -> bf16 conversion (for weight matrices)
// ---------------------------------------------------------------------------
__global__ void f32_to_bf16_kernel(const float* __restrict__ src,
                                   __bf16* __restrict__ dst, int n) {
    int i = blockIdx.x * blockDim.x + threadIdx.x;
    if (i < n) dst[i] = (__bf16)src[i];
}

// ---------------------------------------------------------------------------
// Fused time-shift + lerp: produces k_in, v_in, r_in in bf16
// ---------------------------------------------------------------------------
__global__ void mix_kernel(const float* __restrict__ x,
                           const float* __restrict__ mk,
                           const float* __restrict__ mv,
                           const float* __restrict__ mr,
                           __bf16* __restrict__ kin,
                           __bf16* __restrict__ vin,
                           __bf16* __restrict__ rin) {
    long idx = (long)blockIdx.x * blockDim.x + threadIdx.x;   // < Mtot*Cc
    int  c   = (int)(idx % Cc);
    long row = idx / Cc;
    int  t   = (int)(row % Tt);
    float xv = x[idx];
    float px = (t == 0) ? 0.0f : x[idx - Cc];
    float k = mk[c], v = mv[c], r = mr[c];
    kin[idx] = (__bf16)(xv * k + px * (1.0f - k));
    vin[idx] = (__bf16)(xv * v + px * (1.0f - v));
    rin[idx] = (__bf16)(xv * r + px * (1.0f - r));
}

// ---------------------------------------------------------------------------
// bf16 WMMA GEMM:  Out[m][n] = sum_k A[m][k] * W[n][k]   (fp32 accumulate)
//
// Block: 256 threads = 8 waves stacked in M; block tile 256(M) x 64(N).
// Each wave computes a 32x64 strip: 2 A-frags x 4 B-frags -> 8 WMMAs/k-tile.
// Two k-tiles (2x32) per loop body with explicit ping-pong buffers: buffers
// alternate in place, so no register rotation copies; every fragment load has
// at least one full 8-WMMA block before its consumer's s_wait_loadcnt.
// mode==1 applies sigmoid in the epilogue (receptance path).
// ---------------------------------------------------------------------------
__global__ __launch_bounds__(256)
void gemm_bf16_wmma_kernel(const __bf16* __restrict__ A,
                           const __bf16* __restrict__ W,
                           float* __restrict__ Out,
                           int M, int N, int K, int mode) {
    const int lane = threadIdx.x & 31;
    const int wid  = threadIdx.x >> 5;
    const int half = lane >> 4;     // 0: lanes 0-15, 1: lanes 16-31
    const int l16  = lane & 15;
    const int mrow = blockIdx.x * 256 + wid * 32;
    const int ncol = blockIdx.y * 64;

    // A row pointers (16-bit A 16x32, contiguous k order, same as B):
    // element j -> K = k0 + j + 16*half: one contiguous 16-halfword run
    const __bf16* arow0 = A + (long)(mrow + l16) * K + 16 * half;
    const __bf16* arow1 = arow0 + (long)16 * K;
    // B row pointers (16-bit B 32x16): element j -> K = k0 + j + 16*half
    const __bf16* wrow[4];
#pragma unroll
    for (int nt = 0; nt < 4; ++nt)
        wrow[nt] = W + (long)(ncol + nt * 16 + l16) * K + 16 * half;

    auto loadA = [](const __bf16* p) {
        v8bf lo = *(const v8bf*)(p);        // 16B aligned
        v8bf hi = *(const v8bf*)(p + 8);    // 16B aligned
        v16bf a;
#pragma unroll
        for (int i = 0; i < 8; ++i) { a[i] = lo[i]; a[i + 8] = hi[i]; }
        return a;
    };

    v8f acc[2][4] = {{v8f{}, v8f{}, v8f{}, v8f{}},
                     {v8f{}, v8f{}, v8f{}, v8f{}}};

    auto wmma8 = [&](const v16bf& A0, const v16bf& A1, const v16bf* Bv) {
#pragma unroll
        for (int nt = 0; nt < 4; ++nt) {
            acc[0][nt] = __builtin_amdgcn_wmma_f32_16x16x32_bf16(
                false, A0, false, Bv[nt], (short)0, acc[0][nt], false, false);
            acc[1][nt] = __builtin_amdgcn_wmma_f32_16x16x32_bf16(
                false, A1, false, Bv[nt], (short)0, acc[1][nt], false, false);
        }
    };

    // ---- prologue: load tiles 0 (even buffer) and 1 (odd buffer) ----
    v16bf a0e = loadA(arow0), a1e = loadA(arow1);
    v16bf be[4];
#pragma unroll
    for (int nt = 0; nt < 4; ++nt) be[nt] = *(const v16bf*)(wrow[nt]);
    v16bf a0o = loadA(arow0 + 32), a1o = loadA(arow1 + 32);
    v16bf bo[4];
#pragma unroll
    for (int nt = 0; nt < 4; ++nt) bo[nt] = *(const v16bf*)(wrow[nt] + 32);

    // ---- main loop: 2 k-tiles per iteration, ping-pong in place ----
    // K/32 tiles total (even count); loop consumes tiles k0 and k0+32,
    // refills with tiles k0+64 and k0+96 (always in range for k0 < K-64).
    for (int k0 = 0; k0 < K - 64; k0 += 64) {
        __builtin_prefetch(arow0 + k0 + 256, 0, 3);
        wmma8(a0e, a1e, be);                 // consume even tile k0
        a0e = loadA(arow0 + k0 + 64);        // refill even with tile k0+64
        a1e = loadA(arow1 + k0 + 64);
#pragma unroll
        for (int nt = 0; nt < 4; ++nt)
            be[nt] = *(const v16bf*)(wrow[nt] + k0 + 64);

        wmma8(a0o, a1o, bo);                 // consume odd tile k0+32
        a0o = loadA(arow0 + k0 + 96);        // refill odd with tile k0+96
        a1o = loadA(arow1 + k0 + 96);
#pragma unroll
        for (int nt = 0; nt < 4; ++nt)
            bo[nt] = *(const v16bf*)(wrow[nt] + k0 + 96);
    }
    // ---- tail: buffers hold tiles K-64 and K-32 ----
    wmma8(a0e, a1e, be);
    wmma8(a0o, a1o, bo);

    // ---- store: C/D layout (VGPR r, lane L): N = L%16, M = r + 8*(L/16) ----
#pragma unroll
    for (int mi = 0; mi < 2; ++mi) {
#pragma unroll
        for (int nt = 0; nt < 4; ++nt) {
            const int n = ncol + nt * 16 + l16;
#pragma unroll
            for (int r = 0; r < 8; ++r) {
                const int m = mrow + mi * 16 + r + 8 * half;
                float v = acc[mi][nt][r];
                if (mode == 1) v = 1.0f / (1.0f + __expf(-v));
                Out[(long)m * N + n] = v;
            }
        }
    }
}

// ---------------------------------------------------------------------------
// WKV recurrence (numerically stabilized), sequential over T.
// One thread per (b, c) channel; lanes cover consecutive c -> coalesced.
// Fuses sigmoid(r) * wkv and bf16 conversion for the final GEMM.
// ---------------------------------------------------------------------------
__global__ void wkv_kernel(const float* __restrict__ Kx,
                           const float* __restrict__ Vx,
                           const float* __restrict__ SR,
                           const float* __restrict__ td,
                           const float* __restrict__ tf,
                           __bf16* __restrict__ rwkv) {
    int gid = blockIdx.x * blockDim.x + threadIdx.x;   // 0 .. B*C-1
    int c = gid % Cc;
    int b = gid / Cc;
    float w = -__expf(td[c]);
    float u = tf[c];

    float num = 0.0f, den = 0.0f, mx = -1e38f;
    long base = (long)b * Tt * Cc + c;
    for (int t = 0; t < Tt; ++t) {
        long idx = base + (long)t * Cc;
        float kt = Kx[idx];
        float vt = Vx[idx];
        // output with u-bonus
        float mo = fmaxf(mx, kt + u);
        float e1 = __expf(mx - mo);
        float e2 = __expf(kt + u - mo);
        float out = (e1 * num + e2 * vt) / (e1 * den + e2);
        // state update with decay w
        float mn = fmaxf(mx + w, kt);
        float a1 = __expf(mx + w - mn);
        float a2 = __expf(kt - mn);
        num = a1 * num + a2 * vt;
        den = a1 * den + a2;
        mx = mn;
        rwkv[idx] = (__bf16)(SR[idx] * out);
    }
}

// ---------------------------------------------------------------------------
// Launch
// ---------------------------------------------------------------------------
extern "C" void kernel_launch(void* const* d_in, const int* in_sizes, int n_in,
                              void* d_out, int out_size, void* d_ws, size_t ws_size,
                              hipStream_t stream) {
    (void)in_sizes; (void)n_in; (void)out_size; (void)ws_size;

    const float* x  = (const float*)d_in[0];
    const float* Wk = (const float*)d_in[1];
    const float* Wv = (const float*)d_in[2];
    const float* Wr = (const float*)d_in[3];
    const float* Wo = (const float*)d_in[4];
    const float* td = (const float*)d_in[5];
    const float* tf = (const float*)d_in[6];
    const float* mk = (const float*)d_in[7];
    const float* mv = (const float*)d_in[8];
    const float* mr = (const float*)d_in[9];
    float* out = (float*)d_out;

    // --- workspace layout ---
    char* ws = (char*)d_ws;
    size_t o = 0;
    const size_t wbytes  = (size_t)Cc * Cc * sizeof(__bf16);        // 2 MB
    const size_t actb16  = (size_t)Mtot * Cc * sizeof(__bf16);      // 32 MB
    const size_t actf32  = (size_t)Mtot * Cc * sizeof(float);       // 64 MB
    __bf16* wkb  = (__bf16*)(ws + o); o += wbytes;
    __bf16* wvb  = (__bf16*)(ws + o); o += wbytes;
    __bf16* wrb  = (__bf16*)(ws + o); o += wbytes;
    __bf16* wob  = (__bf16*)(ws + o); o += wbytes;
    __bf16* kin  = (__bf16*)(ws + o); o += actb16;
    __bf16* vin  = (__bf16*)(ws + o); o += actb16;
    __bf16* rin  = (__bf16*)(ws + o); o += actb16;
    float*  Kf   = (float*) (ws + o); o += actf32;
    float*  Vf   = (float*) (ws + o); o += actf32;
    float*  SR   = (float*) (ws + o); o += actf32;
    __bf16* rwkv = (__bf16*)(ws + o); o += actb16;

    // 1) weights -> bf16
    {
        int n = Cc * Cc;
        dim3 g((n + 255) / 256), blk(256);
        f32_to_bf16_kernel<<<g, blk, 0, stream>>>(Wk, wkb, n);
        f32_to_bf16_kernel<<<g, blk, 0, stream>>>(Wv, wvb, n);
        f32_to_bf16_kernel<<<g, blk, 0, stream>>>(Wr, wrb, n);
        f32_to_bf16_kernel<<<g, blk, 0, stream>>>(Wo, wob, n);
    }

    // 2) time-shift + lerp mixes -> bf16 activations
    {
        long n = (long)Mtot * Cc;
        mix_kernel<<<dim3((unsigned)(n / 256)), 256, 0, stream>>>(
            x, mk, mv, mr, kin, vin, rin);
    }

    // 3) projection GEMMs (K, V, R-with-sigmoid)
    {
        dim3 g(Mtot / 256, Cc / 64), blk(256);
        gemm_bf16_wmma_kernel<<<g, blk, 0, stream>>>(kin, wkb, Kf, Mtot, Cc, Cc, 0);
        gemm_bf16_wmma_kernel<<<g, blk, 0, stream>>>(vin, wvb, Vf, Mtot, Cc, Cc, 0);
        gemm_bf16_wmma_kernel<<<g, blk, 0, stream>>>(rin, wrb, SR, Mtot, Cc, Cc, 1);
    }

    // 4) WKV recurrence + sigmoid(r)*wkv -> bf16
    {
        int n = Bb * Cc;   // 8192 channel threads
        wkv_kernel<<<dim3(n / 256), 256, 0, stream>>>(Kf, Vf, SR, td, tf, rwkv);
    }

    // 5) output projection -> d_out (fp32)
    {
        dim3 g(Mtot / 256, Cc / 64), blk(256);
        gemm_bf16_wmma_kernel<<<g, blk, 0, stream>>>(rwkv, wob, out, Mtot, Cc, Cc, 0);
    }
}
